// SinkhornSolver_21517786153373
// MI455X (gfx1250) — matrix operand order repaired and checked by reading, not compiled, not run on hardware
//
#include <hip/hip_runtime.h>

#define BN 8192
#define MAX_ITER 100
#define THRESH_SQ 1e-6f
#define EPSF 1e-8f

typedef __attribute__((ext_vector_type(16))) _Float16 v16h;
typedef __attribute__((ext_vector_type(8)))  _Float16 v8h;
typedef __attribute__((ext_vector_type(2)))  _Float16 v2h;
typedef __attribute__((ext_vector_type(8)))  float    v8f;
typedef __attribute__((ext_vector_type(4)))  float    v4f;
typedef __attribute__((ext_vector_type(2)))  float    v2f;

// ---------------------------------------------------------------------------
// Build K16 = exp(-C/REG) (f16, row-major) and KT16 = transpose(K16).
// C reads and KT16 stores are non-temporal (one-shot / streamed); K16 stores
// are regular so the 128MB K16 image gets resident in the 192MB L2 and stays
// there across all iterations (KT16 is NT-streamed and never evicts it).
// ---------------------------------------------------------------------------
__global__ void __launch_bounds__(256) build_K_kernel(
    const float* __restrict__ C, _Float16* __restrict__ K16,
    _Float16* __restrict__ KT16) {
  __shared__ _Float16 lds[64][66];  // 66-half stride: odd dword -> conflict-free
  const int tx = threadIdx.x;       // 0..31
  const int ty = threadIdx.y;       // 0..7
  const int c2 = tx * 2;
  const long colBase = (long)blockIdx.x * 64;
  const long rowBase = (long)blockIdx.y * 64;
#pragma unroll
  for (int i = 0; i < 8; ++i) {
    const int rl = ty + i * 8;
    const long r = rowBase + rl;
    const v2f cv =
        __builtin_nontemporal_load((const v2f*)(C + r * BN + colBase + c2));
    const _Float16 h0 = (_Float16)__expf(-10.0f * cv[0]);
    const _Float16 h1 = (_Float16)__expf(-10.0f * cv[1]);
    lds[rl][c2] = h0;
    lds[rl][c2 + 1] = h1;
    v2h hv; hv[0] = h0; hv[1] = h1;
    *(v2h*)(K16 + r * BN + colBase + c2) = hv;  // RT store: prefill L2
  }
  __syncthreads();
#pragma unroll
  for (int i = 0; i < 8; ++i) {
    const int pl = ty + i * 8;
    const long p = colBase + pl;  // row of KT
    v2h hv; hv[0] = lds[c2][pl]; hv[1] = lds[c2 + 1][pl];
    __builtin_nontemporal_store(hv, (v2h*)(KT16 + p * BN + rowBase + c2));
  }
}

// ---------------------------------------------------------------------------
// u = v = 1/n, flag = 0
// ---------------------------------------------------------------------------
__global__ void init_uv_kernel(float* __restrict__ u, float* __restrict__ v,
                               int* __restrict__ flag) {
  const int i = blockIdx.x * 256 + threadIdx.x;
  const float inv = 1.0f / (float)BN;
  u[i] = inv;
  v[i] = inv;
  if (i == 0) *flag = 0;
}

// ---------------------------------------------------------------------------
// y[i] = num[i] / (M[i,:] . x + eps), M f16 row-major BNxBN.
// Block = 256 thr = 8 waves, block owns 16 rows, each wave a 1024-wide K
// stripe = 32x v_wmma_f32_16x16x32_f16 with the x-chunk splat into B
// (B layout: lane l holds x at the same 16 K as its A elements; all 16 N columns equal).
// NT=true streams M non-temporally (used for the KT pass so it never evicts
// the L2-resident K16). Constant 32-trip loop -> scalar loop control, EXEC
// stays all-1s through every WMMA.
// ---------------------------------------------------------------------------
template <bool NT>
__global__ void __launch_bounds__(256) sinkhorn_matvec_kernel(
    const _Float16* __restrict__ M, const float* __restrict__ x,
    const float* __restrict__ num, float* __restrict__ y,
    float* __restrict__ blockpart, const int* __restrict__ flag,
    const int do_err) {
  if (*flag) return;  // uniform: converged -> no-op launch

  const int tid = threadIdx.x;
  const int lane = tid & 31;
  const int wave = tid >> 5;
  const int mrow = lane & 15;   // A-matrix: lane%16 = row M
  const int khalf = lane >> 4;  // lanes 16-31 hold the odd K octets
  const long row = (long)blockIdx.x * 16 + mrow;
  const _Float16* rowp = M + row * BN;
  const int k0 = wave * 1024;

  v8f c = {};
#pragma unroll 4
  for (int j = 0; j < 32; ++j) {  // constant trip count -> scalar loop ctrl
    const int kc = k0 + (j << 5);
    const int abase = kc + khalf * 8;
    v8h alo, ahi;
    if (NT) {
      alo = __builtin_nontemporal_load((const v8h*)(rowp + abase));
      ahi = __builtin_nontemporal_load((const v8h*)(rowp + abase + 16));
    } else {
      alo = *(const v8h*)(rowp + abase);
      ahi = *(const v8h*)(rowp + abase + 16);
    }
    const v16h a = __builtin_shufflevector(
        alo, ahi, 0, 1, 2, 3, 4, 5, 6, 7, 8, 9, 10, 11, 12, 13, 14, 15);
    const v8f xlo = *(const v8f*)(x + abase);
    const v8f xhi = *(const v8f*)(x + abase + 16);
    const v16h b = __builtin_convertvector(__builtin_shufflevector(
        xlo, xhi, 0, 1, 2, 3, 4, 5, 6, 7, 8, 9, 10, 11, 12, 13, 14, 15), v16h);
    c = __builtin_amdgcn_wmma_f32_16x16x32_f16(
        false, a, false, b, (short)0, c, false, false);
  }

  // D layout: lane 0 (N=0) holds rows 0-7 in c[0..7]; lane 16 rows 8-15.
  __shared__ float part[8][16];
  __shared__ float dsq[16];
  if (lane == 0) {
#pragma unroll
    for (int r = 0; r < 8; ++r) part[wave][r] = c[r];
  } else if (lane == 16) {
#pragma unroll
    for (int r = 0; r < 8; ++r) part[wave][8 + r] = c[r];
  }
  __syncthreads();

  if (tid < 16) {
    float s = 0.0f;
#pragma unroll
    for (int w = 0; w < 8; ++w) s += part[w][tid];
    const long gr = (long)blockIdx.x * 16 + tid;
    const float ynew = num[gr] / (s + EPSF);
    if (do_err) {
      const float d = ynew - y[gr];
      dsq[tid] = d * d;
    }
    y[gr] = ynew;
  }
  if (do_err) {
    __syncthreads();
    if (tid == 0) {
      float e = 0.0f;
#pragma unroll
      for (int r = 0; r < 16; ++r) e += dsq[r];
      blockpart[blockIdx.x] = e;  // deterministic (no float atomics)
    }
  }
}

// ---------------------------------------------------------------------------
// Deterministic reduction of the 512 per-block error partials; latch flag
// when ||u_new - u_old||^2 < THRESHOLD^2 (matches the while-loop condition).
// ---------------------------------------------------------------------------
__global__ void check_conv_kernel(const float* __restrict__ blockpart,
                                  int* __restrict__ flag) {
  if (*flag) return;
  __shared__ float s[256];
  const int tid = threadIdx.x;
  s[tid] = blockpart[tid] + blockpart[tid + 256];
  __syncthreads();
  for (int off = 128; off > 0; off >>= 1) {
    if (tid < off) s[tid] += s[tid + off];
    __syncthreads();
  }
  if (tid == 0 && s[0] < THRESH_SQ) *flag = 1;
}

// ---------------------------------------------------------------------------
// out[i][j] = u[i] * exp(-C[i][j]/REG) * v[j]. Recomputes exp from pristine C
// (overwrites the f16 scratch living in d_out; nothing here reads it).
// One-shot traffic -> fully non-temporal.
// ---------------------------------------------------------------------------
__global__ void __launch_bounds__(256) finalize_kernel(
    const float* __restrict__ C, const float* __restrict__ u,
    const float* __restrict__ v, float* __restrict__ out) {
  const long row = blockIdx.y;
  const long col = (long)blockIdx.x * 1024 + (long)threadIdx.x * 4;
  const float uu = u[row];
  const v4f cv = __builtin_nontemporal_load((const v4f*)(C + row * BN + col));
  const v4f vv = *(const v4f*)(v + col);
  v4f o;
  o[0] = uu * __expf(-10.0f * cv[0]) * vv[0];
  o[1] = uu * __expf(-10.0f * cv[1]) * vv[1];
  o[2] = uu * __expf(-10.0f * cv[2]) * vv[2];
  o[3] = uu * __expf(-10.0f * cv[3]) * vv[3];
  __builtin_nontemporal_store(o, (v4f*)(out + row * BN + col));
}

extern "C" void kernel_launch(void* const* d_in, const int* in_sizes, int n_in,
                              void* d_out, int out_size, void* d_ws,
                              size_t ws_size, hipStream_t stream) {
  (void)in_sizes; (void)n_in; (void)out_size; (void)ws_size;

  const float* C = (const float*)d_in[0];  // cost_matrix [BN,BN]
  const float* a = (const float*)d_in[1];  // [BN]
  const float* b = (const float*)d_in[2];  // [BN]

  // d_out (BN*BN f32 = 256MB) doubles as scratch for the two f16 matrices
  // (2 x 128MB) during the iterations; finalize overwrites it at the end.
  _Float16* K16 = (_Float16*)d_out;
  _Float16* KT16 = K16 + (size_t)BN * BN;
  float* out = (float*)d_out;

  float* u = (float*)d_ws;      // BN floats
  float* v = u + BN;            // BN floats
  float* blockpart = v + BN;    // 512 floats
  int* flag = (int*)(blockpart + 512);

  build_K_kernel<<<dim3(BN / 64, BN / 64), dim3(32, 8), 0, stream>>>(C, K16,
                                                                     KT16);
  init_uv_kernel<<<BN / 256, 256, 0, stream>>>(u, v, flag);

  for (int it = 0; it < MAX_ITER; ++it) {
    // v = b / (K^T u + eps): KT16 streamed non-temporally (keeps K16 in L2)
    sinkhorn_matvec_kernel<true><<<BN / 16, 256, 0, stream>>>(
        KT16, u, b, v, blockpart, flag, 0);
    // u = a / (K v + eps): K16 via regular RT loads -> L2-resident
    sinkhorn_matvec_kernel<false><<<BN / 16, 256, 0, stream>>>(
        K16, v, a, u, blockpart, flag, 1);
    check_conv_kernel<<<1, 256, 0, stream>>>(blockpart, flag);
  }

  finalize_kernel<<<dim3(BN / 1024, BN), 256, 0, stream>>>(C, u, v, out);
}
